// Spatial_STMoE_38182259261877
// MI455X (gfx1250) — matrix operand order repaired and checked by reading, not compiled, not run on hardware
//
#include <hip/hip_runtime.h>
#include <hip/hip_bf16.h>

// ---- problem constants (from the reference) ----
#define B_    4
#define S_    1024
#define DM_   256
#define H_    8
#define HD_   32
#define CIN_  23
#define COUT_ 256
#define E_    16
#define D2_   512
#define HE_   1365
#define H2_   2730
#define CAP_  128
#define NTOK_ 4096                 // B_*S_
#define NSLOT_ 8192                // E_*B_*CAP_

typedef _Float16 f16;
typedef __attribute__((ext_vector_type(16))) _Float16 v16h;
typedef __attribute__((ext_vector_type(8)))  _Float16 v8h;
typedef __attribute__((ext_vector_type(8)))  float    v8f;

__device__ __forceinline__ float gelu_f(float x) {
  return 0.5f * x * (1.0f + erff(x * 0.70710678118654752f));
}

// ---------------- block reduction helpers (wave32) ----------------
__device__ __forceinline__ float block_reduce_sum(float v, float* sred) {
  for (int o = 16; o > 0; o >>= 1) v += __shfl_xor(v, o, 32);
  int w = threadIdx.x >> 5;
  if ((threadIdx.x & 31) == 0) sred[w] = v;
  __syncthreads();
  float t = 0.0f;
  int nw = blockDim.x >> 5;
  for (int i = 0; i < nw; ++i) t += sred[i];
  __syncthreads();
  return t;
}
__device__ __forceinline__ float block_reduce_max(float v, float* sred) {
  for (int o = 16; o > 0; o >>= 1) v = fmaxf(v, __shfl_xor(v, o, 32));
  int w = threadIdx.x >> 5;
  if ((threadIdx.x & 31) == 0) sred[w] = v;
  __syncthreads();
  float t = -3.4e38f;
  int nw = blockDim.x >> 5;
  for (int i = 0; i < nw; ++i) t = fmaxf(t, sred[i]);
  __syncthreads();
  return t;
}

// ---------------- WMMA tiled GEMM ----------------
// C[M,N] = act(alpha * A[M,K] @ W + bias) + residual
// W is K x N (row major), or with BT, N x K (row major).
// Batched via gridDim.z with flat strides sA/sB/sC/sBias.
// CONTRACT: M is always a multiple of BM (true for every call site here),
// K edge only when KE=1, N edge only when NE=1.
#define BM 64
#define BN 64
#define BK 32
#define LSTR 40   // LDS row stride in f16 (=80B, multiple of 16B -> aligned b128 ds ops)

template <int BT, int KE, int NE>
__global__ __launch_bounds__(256) void wmma_gemm_kernel(
    const float* __restrict__ A, const float* __restrict__ W,
    float* __restrict__ C, const float* __restrict__ bias,
    const float* __restrict__ R,
    int M, int N, int K, int lda, int ldb, int ldc,
    long long sA, long long sB, long long sC, long long sBias,
    float alpha, int act)
{
  __shared__ alignas(16) f16 As[BM][LSTR];
  __shared__ alignas(16) f16 Bs[BN][LSTR];

  const long long z = blockIdx.z;
  A += z * sA;
  W += z * sB;
  C += z * sC;
  if (R)    R    += z * sC;
  if (bias) bias += z * sBias;

  const int tid  = threadIdx.x;
  const int m0   = blockIdx.y * BM;
  const int n0   = blockIdx.x * BN;
  const int wave = tid >> 5, lane = tid & 31;
  const int half = lane >> 4, l16 = lane & 15;
  const int tm   = wave & 3;            // wmma tile row within block (0..3)
  const int tn0  = (wave >> 2) * 2;     // first wmma tile col (0 or 2)

  // staging coordinates (fixed per thread)
  const int ar = tid >> 2, ac = (tid & 3) * 8;            // A: 64 rows x 4 chunks of 8
  const int bn_t = tid >> 2, bc_t = (tid & 3) * 8;        // B (BT): 64 rows x 4 chunks of 8
  const int bk_n = tid >> 3, bn_n = (tid & 7) * 8;        // B (KxN): 32 rows x 8 chunks of 8

  v8f acc0 = {}; v8f acc1 = {};

  for (int k0 = 0; k0 < K; k0 += BK) {
    // ---- stage A tile [BM][BK] (f32 -> f16), one b128 LDS store per thread ----
    {
      const float* ap = A + (long long)(m0 + ar) * lda + (k0 + ac);
      v8h pk;
      if (!KE || (k0 + ac + 8 <= K)) {
        #pragma unroll
        for (int i = 0; i < 8; ++i) pk[i] = (f16)ap[i];
        if (k0 + BK < K) __builtin_prefetch(ap + BK, 0, 1);
      } else {
        #pragma unroll
        for (int i = 0; i < 8; ++i) pk[i] = (f16)((k0 + ac + i < K) ? ap[i] : 0.0f);
      }
      *(v8h*)&As[ar][ac] = pk;
    }
    // ---- stage B tile as Bs[n][k] ----
    if (BT) { // W is N x K: contiguous along k
      const float* bp = W + (long long)(n0 + bn_t) * ldb + (k0 + bc_t);
      const bool nok = !NE || (n0 + bn_t < N);
      v8h pk;
      if (nok && (!KE || (k0 + bc_t + 8 <= K))) {
        #pragma unroll
        for (int i = 0; i < 8; ++i) pk[i] = (f16)bp[i];
        if (k0 + BK < K) __builtin_prefetch(bp + BK, 0, 1);
      } else {
        #pragma unroll
        for (int i = 0; i < 8; ++i)
          pk[i] = (f16)((nok && (k0 + bc_t + i < K)) ? bp[i] : 0.0f);
      }
      *(v8h*)&Bs[bn_t][bc_t] = pk;
    } else {  // W is K x N: contiguous along n, transpose into LDS
      const float* bp = W + (long long)(k0 + bk_n) * ldb + (n0 + bn_n);
      const bool kok = !KE || (k0 + bk_n < K);
      if (!NE) {
        if (kok) {
          f16 tmp[8];
          #pragma unroll
          for (int i = 0; i < 8; ++i) tmp[i] = (f16)bp[i];
          #pragma unroll
          for (int i = 0; i < 8; ++i) Bs[bn_n + i][bk_n] = tmp[i];
          if (k0 + BK < K) __builtin_prefetch(bp + (long long)BK * ldb, 0, 1);
        } else {
          #pragma unroll
          for (int i = 0; i < 8; ++i) Bs[bn_n + i][bk_n] = (f16)0.0f;
        }
      } else {
        #pragma unroll
        for (int i = 0; i < 8; ++i)
          Bs[bn_n + i][bk_n] = (f16)((kok && (n0 + bn_n + i < N)) ? bp[i] : 0.0f);
      }
    }
    __syncthreads();

    // ---- fragments ----
    // A: row m = tm*16 + l16; lane-half h covers K 16h..16h+15
    v8h alo = *(const v8h*)&As[tm * 16 + l16][16 * half];
    v8h ahi = *(const v8h*)&As[tm * 16 + l16][16 * half + 8];
    v16h af = __builtin_shufflevector(alo, ahi, 0,1,2,3,4,5,6,7,8,9,10,11,12,13,14,15);
    // B: col n = tn*16 + l16; lane-half h covers K 16h..16h+15
    v8h b0lo = *(const v8h*)&Bs[tn0 * 16 + l16][16 * half];
    v8h b0hi = *(const v8h*)&Bs[tn0 * 16 + l16][16 * half + 8];
    v16h bf0 = __builtin_shufflevector(b0lo, b0hi, 0,1,2,3,4,5,6,7,8,9,10,11,12,13,14,15);
    v8h b1lo = *(const v8h*)&Bs[(tn0 + 1) * 16 + l16][16 * half];
    v8h b1hi = *(const v8h*)&Bs[(tn0 + 1) * 16 + l16][16 * half + 8];
    v16h bf1 = __builtin_shufflevector(b1lo, b1hi, 0,1,2,3,4,5,6,7,8,9,10,11,12,13,14,15);

    acc0 = __builtin_amdgcn_wmma_f32_16x16x32_f16(false, af, false, bf0, (short)0, acc0, false, false);
    acc1 = __builtin_amdgcn_wmma_f32_16x16x32_f16(false, af, false, bf1, (short)0, acc1, false, false);
    __syncthreads();
  }

  // ---- epilogue: alpha, bias, activation, residual ----
  #pragma unroll
  for (int t = 0; t < 2; ++t) {
    v8f acc = t ? acc1 : acc0;
    int gn = n0 + (tn0 + t) * 16 + l16;
    if (NE && gn >= N) continue;
    float bv = bias ? bias[gn] : 0.0f;
    #pragma unroll
    for (int r = 0; r < 8; ++r) {
      int gm = m0 + tm * 16 + r + 8 * half;   // C layout: VGPR r -> row r (+8 for upper lanes)
      float v = acc[r] * alpha + bv;
      if (act == 1)      v = gelu_f(v);
      else if (act == 2) v = (v > 0.0f) ? v : 0.01f * v;
      if (R) v += R[(long long)gm * ldc + gn];
      C[(long long)gm * ldc + gn] = v;
    }
  }
}

// ---------------- tiny-K projections (VALU; K=3 / K=23 would waste WMMA) ----------------
__global__ void topo_kernel(const float* __restrict__ Kin, const float* __restrict__ Qin,
                            const float* __restrict__ w, const float* __restrict__ b,
                            float* __restrict__ out)
{
  long long t = (long long)blockIdx.x * 256 + threadIdx.x;
  const long long total = (long long)B_ * 2 * S_ * DM_;
  if (t >= total) return;
  int c = (int)(t % DM_);
  long long r = t / DM_;
  int bi = (int)(r / (2 * S_));
  int rr = (int)(r % (2 * S_));
  const float* src = (rr < S_) ? (Kin + ((long long)bi * S_ + rr) * 3)
                               : (Qin + ((long long)bi * S_ + (rr - S_)) * 3);
  float acc = b[c];
  #pragma unroll
  for (int j = 0; j < 3; ++j) acc += src[j] * w[j * DM_ + c];
  out[t] = gelu_f(acc);
}

__global__ void values_kernel(const float* __restrict__ V, const float* __restrict__ w,
                              const float* __restrict__ b, float* __restrict__ out)
{
  long long t = (long long)blockIdx.x * 256 + threadIdx.x;
  if (t >= (long long)NTOK_ * DM_) return;
  int c = (int)(t % DM_);
  long long n = t / DM_;
  const float* src = V + n * CIN_;
  float acc = b[c];
  #pragma unroll
  for (int j = 0; j < CIN_; ++j) acc += src[j] * w[j * DM_ + c];
  out[t] = gelu_f(acc);
}

__global__ void copyq_kernel(const float* __restrict__ topo, float* __restrict__ x)
{
  long long t = (long long)blockIdx.x * 256 + threadIdx.x;
  if (t >= (long long)NTOK_ * DM_) return;
  int j = (int)(t % DM_);
  long long n = t / DM_;               // n = b*S + s
  long long b = n / S_, s = n % S_;
  x[n * D2_ + DM_ + j] = topo[((b * 2 * S_) + S_ + s) * DM_ + j];
}

// ---------------- normalization ----------------
__global__ __launch_bounds__(256) void rmsnorm_kernel(const float* __restrict__ x,
                                                      const float* __restrict__ g,
                                                      float* __restrict__ y)
{
  __shared__ float sred[8];
  long long row = blockIdx.x;
  const float* p = x + row * D2_;
  float s = 0.0f;
  for (int i = threadIdx.x; i < D2_; i += 256) { float v = p[i]; s += v * v; }
  s = block_reduce_sum(s, sred);
  float sc = 22.627416997969522f / fmaxf(sqrtf(s), 1e-12f);  // sqrt(512)/||x||
  for (int i = threadIdx.x; i < D2_; i += 256) y[row * D2_ + i] = p[i] * sc * g[i];
}

__global__ __launch_bounds__(256) void layernorm_expert_kernel(float* __restrict__ x,
                                                               const float* __restrict__ g,
                                                               const float* __restrict__ bb)
{
  __shared__ float sred[8];
  long long row = blockIdx.x;                       // E*B*CAP rows
  int e = (int)(row / (B_ * CAP_));
  float* p = x + row * D2_;
  float s = 0.0f, s2 = 0.0f;
  for (int i = threadIdx.x; i < D2_; i += 256) { float v = p[i]; s += v; s2 += v * v; }
  s  = block_reduce_sum(s, sred);
  s2 = block_reduce_sum(s2, sred);
  float mean = s / (float)D2_;
  float var  = fmaxf(s2 / (float)D2_ - mean * mean, 0.0f);
  float rstd = rsqrtf(var + 1e-5f);
  const float* ge = g  + (long long)e * D2_;
  const float* be = bb + (long long)e * D2_;
  for (int i = threadIdx.x; i < D2_; i += 256)
    p[i] = (p[i] - mean) * rstd * ge[i] + be[i];
}

// ---------------- softmax over rows (attention) ----------------
__global__ __launch_bounds__(256) void softmax_rows_kernel(float* __restrict__ x, int D)
{
  __shared__ float sred[8];
  float* p = x + (long long)blockIdx.x * D;
  float mx = -3.4e38f;
  for (int i = threadIdx.x; i < D; i += 256) mx = fmaxf(mx, p[i]);
  mx = block_reduce_max(mx, sred);
  float s = 0.0f;
  for (int i = threadIdx.x; i < D; i += 256) { float e = expf(p[i] - mx); p[i] = e; s += e; }
  s = block_reduce_sum(s, sred);
  float inv = 1.0f / s;
  for (int i = threadIdx.x; i < D; i += 256) p[i] *= inv;
}

// ---------------- GEGLU gate ----------------
__global__ void glu_kernel(const float* __restrict__ h, const float* __restrict__ mult,
                           float* __restrict__ a)
{
  long long t = (long long)blockIdx.x * 256 + threadIdx.x;
  if (t >= (long long)NTOK_ * HE_) return;
  int i = (int)(t % HE_);
  long long m = t / HE_;
  const float* hr = h + m * H2_;
  a[t] = gelu_f(hr[HE_ + i]) * hr[i] * mult[i];
}

// ---------------- MoE routing ----------------
__global__ void logits_kernel(const float* __restrict__ xr, const float* __restrict__ gw,
                              float* __restrict__ logits)
{
  long long t = (long long)blockIdx.x * 256 + threadIdx.x;
  if (t >= (long long)NTOK_ * E_) return;
  int e = (int)(t % E_);
  long long n = t / E_;
  const float* xp = xr + n * D2_;
  const float* wp = gw + (long long)e * D2_;
  float acc = 0.0f;
  for (int i = 0; i < D2_; ++i) acc += xp[i] * wp[i];
  logits[t] = acc;
}

// stats layout: [0..63] raw_sum(b,e), [64..127] top1 count(b,e), [128] sum(lse^2)
__global__ void routing_token_kernel(const float* __restrict__ logits,
                                     float* __restrict__ wa, float* __restrict__ wb,
                                     int* __restrict__ ia, int* __restrict__ ib,
                                     float* __restrict__ stats)
{
  int n = blockIdx.x * 256 + threadIdx.x;
  if (n >= NTOK_) return;
  const float* lp = logits + (long long)n * E_;
  float raw[E_];
  float mx = -3.4e38f;
  for (int e = 0; e < E_; ++e) { raw[e] = lp[e]; mx = fmaxf(mx, raw[e]); }
  float s = 0.0f;
  for (int e = 0; e < E_; ++e) { raw[e] = expf(raw[e] - mx); s += raw[e]; }
  float inv = 1.0f / s;
  for (int e = 0; e < E_; ++e) raw[e] *= inv;
  float lse = logf(s) + mx;
  atomicAdd(&stats[128], lse * lse);
  int b = n / S_;
  for (int e = 0; e < E_; ++e) atomicAdd(&stats[b * E_ + e], raw[e]);
  int i1 = 0; float v1 = raw[0];
  for (int e = 1; e < E_; ++e) if (raw[e] > v1) { v1 = raw[e]; i1 = e; }
  int i2 = (i1 == 0) ? 1 : 0; float v2 = raw[i2];
  for (int e = 0; e < E_; ++e) { if (e == i1) continue; if (raw[e] > v2) { v2 = raw[e]; i2 = e; } }
  float norm = fmaxf(v1 + v2, 1e-9f);
  float g1 = v1 / norm, g2 = v2 / norm;
  atomicAdd(&stats[64 + b * E_ + i1], 1.0f);
  wa[n] = g1;
  wb[n] = (g2 > 0.2f) ? g2 : 0.0f;   // route2 threshold on normalized gate
  ia[n] = i1; ib[n] = i2;
}

// sequential per-batch capacity scan (exclusive cumsum per expert)
__global__ void route_scan_kernel(const int* __restrict__ ia, const int* __restrict__ ib,
                                  float* __restrict__ wa, float* __restrict__ wb,
                                  int* __restrict__ loc1, int* __restrict__ loc2)
{
  if (threadIdx.x != 0) return;
  int b = blockIdx.x;
  int c1[E_], c2[E_];
  for (int e = 0; e < E_; ++e) { c1[e] = 0; c2[e] = 0; }
  int base = b * S_;
  for (int n = 0; n < S_; ++n) {
    int t = base + n;
    int e = ia[t];
    int p = c1[e]++;
    loc1[t] = (p < CAP_) ? ((e * B_ + b) * CAP_ + p) : -1;
  }
  for (int n = 0; n < S_; ++n) {
    int t = base + n;
    if (wb[t] > 0.0f) {
      int e = ib[t];
      int p = c1[e] + c2[e]++;           // offset by total top1 count for expert
      loc2[t] = (p < CAP_) ? ((e * B_ + b) * CAP_ + p) : -1;
    } else loc2[t] = -1;
    if (loc2[t] < 0) wb[t] = 0.0f;
    if (loc1[t] < 0) wa[t] = 0.0f;
  }
}

__global__ void zero_kernel(float* __restrict__ p, long long n)
{
  long long t = (long long)blockIdx.x * 256 + threadIdx.x;
  if (t < n) p[t] = 0.0f;
}

__global__ void scatter_kernel(const float* __restrict__ xr,
                               const int* __restrict__ loc1, const int* __restrict__ loc2,
                               float* __restrict__ xin)
{
  long long t = (long long)blockIdx.x * 256 + threadIdx.x;
  if (t >= (long long)NTOK_ * D2_) return;
  int d = (int)(t % D2_);
  int n = (int)(t / D2_);
  float v = xr[t];
  int l1 = loc1[n]; if (l1 >= 0) xin[(long long)l1 * D2_ + d] = v;
  int l2 = loc2[n]; if (l2 >= 0) xin[(long long)l2 * D2_ + d] = v;
}

__global__ void combine_kernel(float* __restrict__ x, const float* __restrict__ hout,
                               const float* __restrict__ wa, const float* __restrict__ wb,
                               const int* __restrict__ loc1, const int* __restrict__ loc2)
{
  long long t = (long long)blockIdx.x * 256 + threadIdx.x;
  if (t >= (long long)NTOK_ * D2_) return;
  int d = (int)(t % D2_);
  int n = (int)(t / D2_);
  float v = x[t];
  int l1 = loc1[n]; if (l1 >= 0) v += wa[n] * hout[(long long)l1 * D2_ + d];
  int l2 = loc2[n]; if (l2 >= 0) v += wb[n] * hout[(long long)l2 * D2_ + d];
  x[t] = v;
}

__global__ void finalize_kernel(const float* __restrict__ stats, float* __restrict__ outscal)
{
  if (threadIdx.x != 0 || blockIdx.x != 0) return;
  float bal = 0.0f;
  for (int i = 0; i < B_ * E_; ++i)
    bal += (stats[i] / (float)S_) * (stats[64 + i] / (float)S_);
  bal = bal / (float)(B_ * E_) * (float)(E_ * E_) * 0.01f;
  float z = stats[128] / (float)NTOK_ * 0.001f;
  outscal[0] = bal + z;
  outscal[1] = bal;
  outscal[2] = z;
}

// ---------------- host side ----------------
static void launch_gemm(hipStream_t st, const float* A, const float* Bmat, float* C,
                        const float* bias, const float* R,
                        int M, int N, int K, int lda, int ldb, int ldc,
                        long long sA, long long sB, long long sC, long long sBias,
                        int nz, float alpha, int act, int bt, int ke, int ne)
{
  dim3 grid((N + BN - 1) / BN, (M + BM - 1) / BM, nz);
  dim3 blkd(256);
  if (bt)      wmma_gemm_kernel<1,0,0><<<grid, blkd, 0, st>>>(A, Bmat, C, bias, R, M, N, K, lda, ldb, ldc, sA, sB, sC, sBias, alpha, act);
  else if (ke) wmma_gemm_kernel<0,1,0><<<grid, blkd, 0, st>>>(A, Bmat, C, bias, R, M, N, K, lda, ldb, ldc, sA, sB, sC, sBias, alpha, act);
  else if (ne) wmma_gemm_kernel<0,0,1><<<grid, blkd, 0, st>>>(A, Bmat, C, bias, R, M, N, K, lda, ldb, ldc, sA, sB, sC, sBias, alpha, act);
  else         wmma_gemm_kernel<0,0,0><<<grid, blkd, 0, st>>>(A, Bmat, C, bias, R, M, N, K, lda, ldb, ldc, sA, sB, sC, sBias, alpha, act);
}

static inline unsigned blk(long long n) { return (unsigned)((n + 255) / 256); }

extern "C" void kernel_launch(void* const* d_in, const int* in_sizes, int n_in,
                              void* d_out, int out_size, void* d_ws, size_t ws_size,
                              hipStream_t stream)
{
  (void)in_sizes; (void)n_in; (void)out_size; (void)ws_size;
  const float* Kin    = (const float*)d_in[0];
  const float* Vin    = (const float*)d_in[1];
  const float* Qin    = (const float*)d_in[2];
  const float* w_topo = (const float*)d_in[3];
  const float* b_topo = (const float*)d_in[4];
  const float* w_val  = (const float*)d_in[5];
  const float* b_val  = (const float*)d_in[6];
  const float* wq = (const float*)d_in[7],  *bq = (const float*)d_in[8];
  const float* wk = (const float*)d_in[9],  *bk = (const float*)d_in[10];
  const float* wv = (const float*)d_in[11], *bv = (const float*)d_in[12];
  const float* wo = (const float*)d_in[13], *bo = (const float*)d_in[14];
  const float* ffb_g   = (const float*)d_in[15];
  const float* ffb_w1  = (const float*)d_in[16];
  const float* ffb_b1  = (const float*)d_in[17];
  const float* ffb_mult= (const float*)d_in[18];
  const float* ffb_w2  = (const float*)d_in[19];
  const float* ffb_b2  = (const float*)d_in[20];
  const float* prenorm_g = (const float*)d_in[21];
  const float* gate_w  = (const float*)d_in[22];
  const float* exp_ln_g= (const float*)d_in[23];
  const float* exp_ln_b= (const float*)d_in[24];
  const float* exp_w1  = (const float*)d_in[25];
  const float* exp_b1  = (const float*)d_in[26];
  const float* exp_w2  = (const float*)d_in[27];
  const float* exp_b2  = (const float*)d_in[28];
  const float* ffa_g   = (const float*)d_in[29];
  const float* ffa_w1  = (const float*)d_in[30];
  const float* ffa_b1  = (const float*)d_in[31];
  const float* ffa_mult= (const float*)d_in[32];
  const float* ffa_w2  = (const float*)d_in[33];
  const float* ffa_b2  = (const float*)d_in[34];
  const float* w_out   = (const float*)d_in[35];
  const float* b_out   = (const float*)d_in[36];
  float* out = (float*)d_out;

  // workspace carve-up (floats)
  float* wsf = (float*)d_ws;
  size_t off = 0;
  auto alloc = [&](size_t n) { float* p = wsf + off; off += n; return p; };
  float* topo   = alloc((size_t)B_ * 2 * S_ * DM_);   // keys | queries
  float* vals   = alloc((size_t)NTOK_ * DM_);
  float* qb     = alloc((size_t)NTOK_ * DM_);
  float* kb     = alloc((size_t)NTOK_ * DM_);
  float* vb     = alloc((size_t)NTOK_ * DM_);
  float* ob     = alloc((size_t)NTOK_ * DM_);
  float* x      = alloc((size_t)NTOK_ * D2_);
  float* xr     = alloc((size_t)NTOK_ * D2_);
  float* hbuf   = alloc((size_t)NTOK_ * H2_);         // also expert hidden (same size)
  float* abuf   = alloc((size_t)NTOK_ * HE_);
  float* scores = alloc((size_t)H_ * S_ * S_);        // per-batch scores; reused below
  float* xin    = scores;                             // E*B*CAP x 512
  float* hout   = scores + (size_t)NSLOT_ * D2_;      // E*B*CAP x 512
  float* logits = alloc((size_t)NTOK_ * E_);
  float* wa     = alloc(NTOK_);
  float* wb2    = alloc(NTOK_);
  int*   ia     = (int*)alloc(NTOK_);
  int*   ib     = (int*)alloc(NTOK_);
  int*   loc1   = (int*)alloc(NTOK_);
  int*   loc2   = (int*)alloc(NTOK_);
  float* stats  = alloc(129);

  zero_kernel<<<1, 256, 0, stream>>>(stats, 129);

  // ---- input projections (tiny K -> VALU) ----
  topo_kernel<<<blk((long long)B_ * 2 * S_ * DM_), 256, 0, stream>>>(Kin, Qin, w_topo, b_topo, topo);
  values_kernel<<<blk((long long)NTOK_ * DM_), 256, 0, stream>>>(Vin, w_val, b_val, vals);
  copyq_kernel<<<blk((long long)NTOK_ * DM_), 256, 0, stream>>>(topo, x);  // x[:,256:] = queries

  // ---- q/k/v projections (WMMA, no edges) ----
  launch_gemm(stream, topo + (size_t)S_ * DM_, wq, qb, bq, nullptr,
              S_, DM_, DM_, DM_, DM_, DM_,
              (long long)2 * S_ * DM_, 0, (long long)S_ * DM_, 0, B_, 1.0f, 0, 0, 0, 0);
  launch_gemm(stream, topo, wk, kb, bk, nullptr,
              S_, DM_, DM_, DM_, DM_, DM_,
              (long long)2 * S_ * DM_, 0, (long long)S_ * DM_, 0, B_, 1.0f, 0, 0, 0, 0);
  launch_gemm(stream, vals, wv, vb, bv, nullptr,
              NTOK_, DM_, DM_, DM_, DM_, DM_, 0, 0, 0, 0, 1, 1.0f, 0, 0, 0, 0);

  // ---- attention, one batch at a time (8 heads batched in gridDim.z) ----
  const float scale = 0.17677669529663687f; // 1/sqrt(32)
  for (int b = 0; b < B_; ++b) {
    launch_gemm(stream, qb + (size_t)b * S_ * DM_, kb + (size_t)b * S_ * DM_, scores,
                nullptr, nullptr,
                S_, S_, HD_, DM_, DM_, S_,
                HD_, HD_, (long long)S_ * S_, 0, H_, scale, 0, 1, 0, 0);   // BT
    softmax_rows_kernel<<<H_ * S_, 256, 0, stream>>>(scores, S_);
    launch_gemm(stream, scores, vb + (size_t)b * S_ * DM_, ob + (size_t)b * S_ * DM_,
                nullptr, nullptr,
                S_, HD_, S_, S_, DM_, DM_,
                (long long)S_ * S_, HD_, HD_, 0, H_, 1.0f, 0, 0, 0, 1);    // N=32 edge
  }

  // o @ wo + bo -> x[:, :256]
  launch_gemm(stream, ob, wo, x, bo, nullptr,
              NTOK_, DM_, DM_, DM_, DM_, D2_, 0, 0, 0, 0, 1, 1.0f, 0, 0, 0, 0);

  // ---- GEGLU ffb (+residual) ----
  rmsnorm_kernel<<<NTOK_, 256, 0, stream>>>(x, ffb_g, xr);
  launch_gemm(stream, xr, ffb_w1, hbuf, ffb_b1, nullptr,
              NTOK_, H2_, D2_, D2_, H2_, H2_, 0, 0, 0, 0, 1, 1.0f, 0, 0, 0, 1);  // N=2730 edge
  glu_kernel<<<blk((long long)NTOK_ * HE_), 256, 0, stream>>>(hbuf, ffb_mult, abuf);
  launch_gemm(stream, abuf, ffb_w2, x, ffb_b2, x,
              NTOK_, D2_, HE_, HE_, D2_, D2_, 0, 0, 0, 0, 1, 1.0f, 0, 0, 1, 0);  // K=1365 edge

  // ---- MoE ----
  rmsnorm_kernel<<<NTOK_, 256, 0, stream>>>(x, prenorm_g, xr);
  logits_kernel<<<blk((long long)NTOK_ * E_), 256, 0, stream>>>(xr, gate_w, logits);
  routing_token_kernel<<<blk(NTOK_), 256, 0, stream>>>(logits, wa, wb2, ia, ib, stats);
  route_scan_kernel<<<B_, 1, 0, stream>>>(ia, ib, wa, wb2, loc1, loc2);
  zero_kernel<<<blk((long long)NSLOT_ * D2_), 256, 0, stream>>>(xin, (long long)NSLOT_ * D2_);
  scatter_kernel<<<blk((long long)NTOK_ * D2_), 256, 0, stream>>>(xr, loc1, loc2, xin);
  layernorm_expert_kernel<<<NSLOT_, 256, 0, stream>>>(xin, exp_ln_g, exp_ln_b);
  launch_gemm(stream, xin, exp_w1, hbuf, exp_b1, nullptr,
              B_ * CAP_, HE_, D2_, D2_, HE_, HE_,
              (long long)B_ * CAP_ * D2_, (long long)D2_ * HE_,
              (long long)B_ * CAP_ * HE_, HE_, E_, 1.0f, 2, 0, 0, 1);            // leaky, N=1365 edge
  launch_gemm(stream, hbuf, exp_w2, hout, exp_b2, nullptr,
              B_ * CAP_, D2_, HE_, HE_, D2_, D2_,
              (long long)B_ * CAP_ * HE_, (long long)HE_ * D2_,
              (long long)B_ * CAP_ * D2_, D2_, E_, 1.0f, 0, 0, 1, 0);            // K=1365 edge
  combine_kernel<<<blk((long long)NTOK_ * D2_), 256, 0, stream>>>(x, hout, wa, wb2, loc1, loc2);

  // ---- GEGLU ffa (+residual) ----
  rmsnorm_kernel<<<NTOK_, 256, 0, stream>>>(x, ffa_g, xr);
  launch_gemm(stream, xr, ffa_w1, hbuf, ffa_b1, nullptr,
              NTOK_, H2_, D2_, D2_, H2_, H2_, 0, 0, 0, 0, 1, 1.0f, 0, 0, 0, 1);  // N=2730 edge
  glu_kernel<<<blk((long long)NTOK_ * HE_), 256, 0, stream>>>(hbuf, ffa_mult, abuf);
  launch_gemm(stream, abuf, ffa_w2, x, ffa_b2, x,
              NTOK_, D2_, HE_, HE_, D2_, D2_, 0, 0, 0, 0, 1, 1.0f, 0, 0, 1, 0);  // K=1365 edge

  // ---- output projection + aux losses ----
  launch_gemm(stream, x, w_out, out, b_out, nullptr,
              NTOK_, COUT_, D2_, D2_, COUT_, COUT_, 0, 0, 0, 0, 1, 1.0f, 1, 0, 0, 0); // gelu
  finalize_kernel<<<1, 1, 0, stream>>>(stats, out + (size_t)NTOK_ * COUT_);
}
